// TernaryLinear_47742856462736
// MI455X (gfx1250) — matrix operand order repaired and checked by reading, not compiled, not run on hardware
//
#include <hip/hip_runtime.h>
#include <hip/hip_bf16.h>
#include <stdint.h>

// Problem dimensions (from reference): x [4,2048,2048] f32, w [8192,2048] f32
#define IN_F    2048
#define OUT_F   8192
#define BATCH_M 8192                       // 4*2048 flattened rows of x
#define WCOUNT  (OUT_F * IN_F)             // 16,777,216
#define XCOUNT  (BATCH_M * IN_F)           // 16,777,216

typedef __attribute__((ext_vector_type(16))) __bf16 bf16x16;
typedef __attribute__((ext_vector_type(8)))  float  f32x8;

static __device__ __forceinline__ uint16_t f32_to_bf16_rne(float f) {
    union { float f; uint32_t u; } c; c.f = f;
    uint32_t u = c.u;
    u += 0x7fffu + ((u >> 16) & 1u);       // round-to-nearest-even
    return (uint16_t)(u >> 16);
}

// ---------------- scale = mean(|w|) + 1e-8 ----------------

__global__ void k_zero(float* s) {
    if (threadIdx.x == 0 && blockIdx.x == 0) s[0] = 0.0f;
}

__global__ void __launch_bounds__(256) k_absreduce(const float* __restrict__ w,
                                                   float* __restrict__ s) {
    __shared__ float red[256];
    float acc = 0.0f;
    for (int i = blockIdx.x * 256 + threadIdx.x; i < WCOUNT; i += gridDim.x * 256)
        acc += fabsf(w[i]);
    red[threadIdx.x] = acc;
    __syncthreads();
    for (int off = 128; off > 0; off >>= 1) {
        if (threadIdx.x < off) red[threadIdx.x] += red[threadIdx.x + off];
        __syncthreads();
    }
    if (threadIdx.x == 0) atomicAdd(s, red[0]);
}

__global__ void k_finalize(float* s) {
    if (threadIdx.x == 0 && blockIdx.x == 0)
        s[1] = s[0] / (float)WCOUNT + 1e-8f;
}

// ---------------- quantize w -> ternary bf16 {-1,0,1} (exact) ----------------

__global__ void __launch_bounds__(256) k_quant(const float* __restrict__ w,
                                               const float* __restrict__ s,
                                               uint16_t* __restrict__ wq) {
    const float inv = 1.0f / s[1];
    for (int i = blockIdx.x * 256 + threadIdx.x; i < WCOUNT; i += gridDim.x * 256) {
        float q = rintf(w[i] * inv);
        q = fminf(1.0f, fmaxf(-1.0f, q));
        wq[i] = f32_to_bf16_rne(q);        // exact for {-1,0,1}
    }
}

// ---------------- convert x -> bf16 ----------------

__global__ void __launch_bounds__(256) k_cvt(const float* __restrict__ x,
                                             uint16_t* __restrict__ xq) {
    for (int i = blockIdx.x * 256 + threadIdx.x; i < XCOUNT; i += gridDim.x * 256)
        xq[i] = f32_to_bf16_rne(x[i]);
}

// ---------------- WMMA bf16 GEMM: out = (Xq @ Wq^T) * scale ----------------
// Block: 256 threads = 8 waves as 2(M) x 4(N).  Wave tile: 64x64 = 4x4 WMMA tiles.
// Fragment layouts per CDNA5 ISA §7.12.2 (wave32):
//   A 16x32 bf16: lane(0..15)=row, kh=lane>>4; elems[0..7]=K[kh*8..+7], elems[8..15]=K[16+kh*8..+7]
//   B 32x16 bf16: lane&15=col,     kh=lane>>4; elems[0..7]=K[kh*8..+7], elems[8..15]=K[16+kh*8..+7]
//   C/D 16x16 f32: elem r -> M = r + 8*kh, N = lane&15
// Both operands are K-contiguous in memory -> direct global_load_b128, no LDS.

union Frag { uint4 u4[2]; bf16x16 v; };

__global__ void __launch_bounds__(256) k_gemm(const uint16_t* __restrict__ Xq,
                                              const uint16_t* __restrict__ Wq,
                                              const float* __restrict__ s,
                                              float* __restrict__ out) {
    const int lane = threadIdx.x & 31;
    const int wave = threadIdx.x >> 5;
    const int wm   = wave >> 2;                        // 0..1
    const int wn   = wave & 3;                         // 0..3
    const int l16  = lane & 15;
    const int kh   = lane >> 4;                        // 0 or 1

    const int row_base = blockIdx.y * 128 + wm * 64;
    const int col_base = blockIdx.x * 256 + wn * 64;

    // per-lane base pointers in 8-bf16 (16B) chunk units
    const uint4* xrow[4];
    const uint4* wcol[4];
#pragma unroll
    for (int t = 0; t < 4; t++) {
        xrow[t] = (const uint4*)(Xq + (size_t)(row_base + t * 16 + l16) * IN_F);
        wcol[t] = (const uint4*)(Wq + (size_t)(col_base + t * 16 + l16) * IN_F);
    }

    f32x8 acc[4][4] = {};

    for (int k0 = 0; k0 < IN_F; k0 += 32) {
        const int c = k0 >> 3;                         // 16B-chunk index
        Frag a[4], b[4];
#pragma unroll
        for (int t = 0; t < 4; t++) {
            a[t].u4[0] = xrow[t][c + kh];              // K = k0 + kh*8 .. +7
            a[t].u4[1] = xrow[t][c + 2 + kh];          // K = k0+16 + kh*8 .. +7
            b[t].u4[0] = wcol[t][c + kh];              // K = k0 + kh*8 .. +7
            b[t].u4[1] = wcol[t][c + 2 + kh];          // K = k0+16 + kh*8 .. +7
        }
#pragma unroll
        for (int mt = 0; mt < 4; mt++) {
#pragma unroll
            for (int nt = 0; nt < 4; nt++) {
                acc[mt][nt] = __builtin_amdgcn_wmma_f32_16x16x32_bf16(
                    /*neg_a=*/false, a[mt].v,
                    /*neg_b=*/false, b[nt].v,
                    /*c_mod=*/(short)0, acc[mt][nt],
                    /*reuse_a=*/false, /*reuse_b=*/false);
            }
        }
    }

    const float scale = s[1];
#pragma unroll
    for (int mt = 0; mt < 4; mt++) {
#pragma unroll
        for (int nt = 0; nt < 4; nt++) {
            const int col = col_base + nt * 16 + l16;
#pragma unroll
            for (int r = 0; r < 8; r++) {
                const int row = row_base + mt * 16 + kh * 8 + r;
                out[(size_t)row * OUT_F + col] = acc[mt][nt][r] * scale;
            }
        }
    }
}

// ---------------- launcher ----------------

extern "C" void kernel_launch(void* const* d_in, const int* in_sizes, int n_in,
                              void* d_out, int out_size, void* d_ws, size_t ws_size,
                              hipStream_t stream) {
    const float* x = (const float*)d_in[0];   // [4,2048,2048]
    const float* w = (const float*)d_in[1];   // [8192,2048]
    float* out     = (float*)d_out;           // [4,2048,8192]

    // workspace layout: [0..1] f32 sum/scale | xq bf16 (32MB) | wq bf16 (32MB)
    float*    s  = (float*)d_ws;
    uint16_t* xq = (uint16_t*)((char*)d_ws + 256);
    uint16_t* wq = (uint16_t*)((char*)d_ws + 256 + (size_t)XCOUNT * 2);

    k_zero     <<<1,    1,   0, stream>>>(s);
    k_absreduce<<<1024, 256, 0, stream>>>(w, s);
    k_finalize <<<1,    1,   0, stream>>>(s);
    k_quant    <<<2048, 256, 0, stream>>>(w, s, wq);
    k_cvt      <<<2048, 256, 0, stream>>>(x, xq);

    dim3 grid(OUT_F / 256, BATCH_M / 128);    // (32, 64)
    k_gemm<<<grid, 256, 0, stream>>>(xq, wq, s, out);
}
